// Actor_59365037965882
// MI455X (gfx1250) — matrix operand order repaired and checked by reading, not compiled, not run on hardware
//
#include <hip/hip_runtime.h>

#define NHEAD  8
#define DMODEL 128
#define DHEAD  16
#define FFDIM  512
#define NFREQ  10
#define NLAYER 3
#define LN_EPS 1e-5f

typedef __bf16 bf16_t;
typedef __attribute__((ext_vector_type(16))) __bf16 v16bf;
typedef __attribute__((ext_vector_type(8)))  float  v8f;

// ---------------------------------------------------------------------------
// Fragment-major packing (CDNA5 ISA 7.12.2, wave32).
// A 16x32 bf16 fragment: lane L holds M=L&15; K half offset 8 if L>=16;
//   vgpr pair g -> K = (g<4 ? 2g : 16+2(g-4)) (+odd).
// B 32x16 bf16 fragment: lane L holds N=L&15; K half offset 8 if L>=16; pair g -> same K as A.
// Packed storage: frag_buf[((tile)*32 + lane)*16 + j]  => per-lane 32 contiguous
// bytes, loadable as one v16bf (2 x b128).
// ---------------------------------------------------------------------------
__device__ __forceinline__ int a_pat(int j) {           // j in [0,16)
  int g = j >> 1, odd = j & 1;
  return ((g < 4) ? (2 * g) : (16 + 2 * (g - 4))) + odd;
}

// pack row-major [K][NCOLS] f32 weight into B-fragment-major bf16
__global__ void pack_B_kernel(const float* __restrict__ W, bf16_t* __restrict__ out,
                              int K, int NCOLS) {
  int t = blockIdx.x * blockDim.x + threadIdx.x;
  if (t >= K * NCOLS) return;
  int j    = t & 15;
  int lane = (t >> 4) & 31;
  int tile = t >> 9;               // kbi * (NCOLS/16) + nt
  int ntn  = NCOLS >> 4;
  int kbi  = tile / ntn, nt = tile - kbi * ntn;
  int g = j >> 1, odd = j & 1;
  int k = kbi * 32 + ((lane & 16) ? 8 : 0) + ((g < 4) ? (2 * g) : (16 + 2 * (g - 4))) + odd;  // table order, same as A
  int n = nt * 16 + (lane & 15);
  out[t] = (bf16_t)W[k * NCOLS + n];
}

__device__ __forceinline__ v16bf load_frag_gl(const bf16_t* __restrict__ Wp,
                                              int kbi, int nt, int ntn, int lane) {
  return *(const v16bf*)(Wp + ((((kbi * ntn + nt) * 32) + lane) << 4));
}
__device__ __forceinline__ v16bf load_frag_lds(const bf16_t* lds, int kbi, int lane) {
  return *(const v16bf*)(lds + (((kbi * 32) + lane) << 4));
}

// ---------------------------------------------------------------------------
// monotone float<->uint encoding for atomicMax-based segment max
// ---------------------------------------------------------------------------
__device__ __forceinline__ unsigned enc_f32(float f) {
  unsigned u = __float_as_uint(f);
  return (u & 0x80000000u) ? ~u : (u | 0x80000000u);
}
__device__ __forceinline__ float dec_f32(unsigned u) {
  unsigned b = (u & 0x80000000u) ? (u & 0x7fffffffu) : ~u;
  return __uint_as_float(b);
}

// ---------------------------------------------------------------------------
// input projections: inp = fa@W_in + b_in ; x = npa@W_emb + b_emb
// ---------------------------------------------------------------------------
__global__ void proj_in_kernel(const float* __restrict__ fa, const float* __restrict__ npa,
                               const float* __restrict__ Win, const float* __restrict__ bin,
                               const float* __restrict__ Wemb, const float* __restrict__ bemb,
                               float* __restrict__ inp, float* __restrict__ x, int N) {
  int t = blockIdx.x * blockDim.x + threadIdx.x;
  if (t >= N * DMODEL) return;
  int n = t / DMODEL, d = t % DMODEL;
  float a = bin[d];
#pragma unroll
  for (int j = 0; j < NFREQ; ++j) a += fa[n * NFREQ + j] * Win[j * DMODEL + d];
  inp[t] = a;
  float b = bemb[d];
#pragma unroll
  for (int j = 0; j < 8; ++j) b += npa[n * 8 + j] * Wemb[j * DMODEL + d];
  x[t] = b;
}

__global__ void add_kernel(float* __restrict__ x, const float* __restrict__ y, int n) {
  int t = blockIdx.x * blockDim.x + threadIdx.x;
  if (t < n) x[t] += y[t];
}

__global__ void attn_init_kernel(unsigned* __restrict__ mb, float* __restrict__ sb, int n) {
  int t = blockIdx.x * blockDim.x + threadIdx.x;
  if (t < n) { mb[t] = 0x007fffffu /* enc(-inf) */; sb[t] = 0.f; }
}

// ---------------------------------------------------------------------------
// QKV + skip: block = 8 waves per 16-row tile; wave w owns column tile w.
// x tile converted+swizzled to A-fragment order in LDS; A frags reused across
// the 4 weight matrices; results staged in LDS, stored coalesced as float4.
// ---------------------------------------------------------------------------
__global__ __launch_bounds__(256)
void qkvs_wmma_kernel(const float* __restrict__ x,
                      const bf16_t* __restrict__ wq, const bf16_t* __restrict__ wk,
                      const bf16_t* __restrict__ wv, const bf16_t* __restrict__ wsk,
                      const float* __restrict__ bq, const float* __restrict__ bk,
                      const float* __restrict__ bv, const float* __restrict__ bsk,
                      float* __restrict__ qo, float* __restrict__ ko,
                      float* __restrict__ vo, float* __restrict__ so) {
  __shared__ bf16_t xt[16 * DMODEL];          //  4 KB, A-fragment-major
  __shared__ float  sh[4 * 16 * DMODEL];      // 32 KB, row-major result tiles
  int row0 = blockIdx.x * 16;
  for (int i = threadIdx.x; i < 16 * DMODEL; i += 256) {
    int j = i & 15, lane = (i >> 4) & 31, kbi = i >> 9;
    int m = lane & 15;
    int k = kbi * 32 + ((lane & 16) ? 8 : 0) + a_pat(j);
    xt[i] = (bf16_t)x[(row0 + m) * DMODEL + k];
  }
  __syncthreads();

  int wave = threadIdx.x >> 5, lane = threadIdx.x & 31;
  v16bf afrag[4];
#pragma unroll
  for (int kb = 0; kb < 4; ++kb) afrag[kb] = load_frag_lds(xt, kb, lane);

  const bf16_t* W[4] = {wq, wk, wv, wsk};
  const float*  B[4] = {bq, bk, bv, bsk};
  int n  = wave * 16 + (lane & 15);
  int mo = (lane & 16) ? 8 : 0;
#pragma unroll
  for (int w = 0; w < 4; ++w) {
    v8f acc = {};
#pragma unroll
    for (int kb = 0; kb < 4; ++kb) {
      v16bf bfrag = load_frag_gl(W[w], kb, wave, DMODEL / 16, lane);
      acc = __builtin_amdgcn_wmma_f32_16x16x32_bf16(false, afrag[kb], false, bfrag,
                                                    (short)0, acc, false, false);
    }
    float bias = B[w][n];
#pragma unroll
    for (int g = 0; g < 8; ++g)
      sh[w * 16 * DMODEL + (mo + g) * DMODEL + n] = acc[g] + bias;
  }
  __syncthreads();

  float* O[4] = {qo, ko, vo, so};
#pragma unroll
  for (int w = 0; w < 4; ++w) {
    float4* dst = (float4*)(O[w] + (size_t)row0 * DMODEL);
    const float4* src = (const float4*)(sh + w * 16 * DMODEL);
    for (int i = threadIdx.x; i < 16 * DMODEL / 4; i += 256) dst[i] = src[i];
  }
}

// ---------------------------------------------------------------------------
// edge kernels: alpha+segmax, exp+segsum, normalized message scatter
// We (8x128) staged in LDS; all per-edge rows loaded as float4 (b128)
// ---------------------------------------------------------------------------
__global__ __launch_bounds__(256)
void attn_alpha_kernel(const float* __restrict__ q, const float* __restrict__ k,
                       const float* __restrict__ epa, const float* __restrict__ We,
                       const int* __restrict__ ei,
                       float* __restrict__ alpha, unsigned* __restrict__ mb, int E) {
  __shared__ float Wes[8 * DMODEL];
  for (int i = threadIdx.x; i < 8 * DMODEL; i += 256) Wes[i] = We[i];
  __syncthreads();
  int t = blockIdx.x * blockDim.x + threadIdx.x;
  if (t >= E * NHEAD) return;
  int e = t >> 3, h = t & 7;
  int src = ei[e], dst = ei[E + e];
  float4 ep4[2];
  ep4[0] = ((const float4*)(epa + (size_t)e * 8))[0];
  ep4[1] = ((const float4*)(epa + (size_t)e * 8))[1];
  const float* ep = (const float*)ep4;
  float4 qa[4], ka[4];
  const float4* qr = (const float4*)(q + (size_t)dst * DMODEL + h * DHEAD);
  const float4* kr = (const float4*)(k + (size_t)src * DMODEL + h * DHEAD);
#pragma unroll
  for (int r = 0; r < 4; ++r) { qa[r] = qr[r]; ka[r] = kr[r]; }
  const float* qv = (const float*)qa;
  const float* kv = (const float*)ka;
  float acc = 0.f;
#pragma unroll
  for (int d = 0; d < DHEAD; ++d) {
    float ev = 0.f;
#pragma unroll
    for (int j = 0; j < 8; ++j) ev += ep[j] * Wes[j * DMODEL + h * DHEAD + d];
    acc += qv[d] * (kv[d] + ev);
  }
  acc *= 0.25f;  // 1/sqrt(16)
  alpha[t] = acc;
  atomicMax(&mb[dst * NHEAD + h], enc_f32(acc));
}

__global__ void attn_exp_kernel(float* __restrict__ alpha, const unsigned* __restrict__ mb,
                                float* __restrict__ sb, const int* __restrict__ ei, int E) {
  int t = blockIdx.x * blockDim.x + threadIdx.x;
  if (t >= E * NHEAD) return;
  int e = t >> 3, h = t & 7;
  int dst = ei[E + e];
  float m = dec_f32(mb[dst * NHEAD + h]);
  float p = __expf(alpha[t] - m);
  alpha[t] = p;
  atomicAdd(&sb[dst * NHEAD + h], p);
}

__global__ __launch_bounds__(256)
void attn_msg_kernel(const float* __restrict__ alpha, const float* __restrict__ sb,
                     const float* __restrict__ v, const float* __restrict__ epa,
                     const float* __restrict__ We, const int* __restrict__ ei,
                     float* __restrict__ msg, int E) {
  __shared__ float Wes[8 * DMODEL];
  for (int i = threadIdx.x; i < 8 * DMODEL; i += 256) Wes[i] = We[i];
  __syncthreads();
  int t = blockIdx.x * blockDim.x + threadIdx.x;
  if (t >= E * NHEAD) return;
  int e = t >> 3, h = t & 7;
  int src = ei[e], dst = ei[E + e];
  float s = sb[dst * NHEAD + h];
  float a = alpha[t] / (s > 0.f ? s : 1.f);
  float4 ep4[2];
  ep4[0] = ((const float4*)(epa + (size_t)e * 8))[0];
  ep4[1] = ((const float4*)(epa + (size_t)e * 8))[1];
  const float* ep = (const float*)ep4;
  float4 va[4];
  const float4* vr = (const float4*)(v + (size_t)src * DMODEL + h * DHEAD);
#pragma unroll
  for (int r = 0; r < 4; ++r) va[r] = vr[r];
  const float* vv = (const float*)va;
  float* mr = msg + (size_t)dst * DMODEL + h * DHEAD;
#pragma unroll
  for (int d = 0; d < DHEAD; ++d) {
    float ev = 0.f;
#pragma unroll
    for (int j = 0; j < 8; ++j) ev += ep[j] * Wes[j * DMODEL + h * DHEAD + d];
    atomicAdd(&mr[d], a * (vv[d] + ev));
  }
}

// ---------------------------------------------------------------------------
// x = LN(x + msg) * g1 + beta1   (one warp per row, lane covers 4 strided cols)
// ---------------------------------------------------------------------------
__global__ void postattn_ln_kernel(float* __restrict__ x, const float* __restrict__ msg,
                                   const float* __restrict__ g1, const float* __restrict__ be1,
                                   int N) {
  int warp = (blockIdx.x * blockDim.x + threadIdx.x) >> 5;
  int lane = threadIdx.x & 31;
  if (warp >= N) return;
  float tv[4], s = 0.f, ss = 0.f;
#pragma unroll
  for (int c0 = 0; c0 < 4; ++c0) {
    int c = c0 * 32 + lane;
    float t = x[(size_t)warp * DMODEL + c] + msg[(size_t)warp * DMODEL + c];
    tv[c0] = t; s += t; ss += t * t;
  }
#pragma unroll
  for (int m = 1; m < 32; m <<= 1) { s += __shfl_xor(s, m, 32); ss += __shfl_xor(ss, m, 32); }
  float mu = s * (1.f / DMODEL);
  float var = ss * (1.f / DMODEL) - mu * mu;
  float rstd = rsqrtf(var + LN_EPS);
#pragma unroll
  for (int c0 = 0; c0 < 4; ++c0) {
    int c = c0 * 32 + lane;
    x[(size_t)warp * DMODEL + c] = (tv[c0] - mu) * rstd * g1[c] + be1[c];
  }
}

// ---------------------------------------------------------------------------
// fused FFN + residual + LN.  h-tile written to LDS directly in A-fragment
// order so stage-2 fragment loads are contiguous ds_load_b128s.
// ---------------------------------------------------------------------------
__global__ __launch_bounds__(256)
void ffn_wmma_kernel(float* __restrict__ x,
                     const bf16_t* __restrict__ w1, const float* __restrict__ b1,
                     const bf16_t* __restrict__ w2, const float* __restrict__ b2,
                     const float* __restrict__ g2, const float* __restrict__ be2) {
  __shared__ bf16_t xt[16 * DMODEL];   //  4 KB, A-frag-major
  __shared__ bf16_t ht[16 * FFDIM];    // 16 KB, A-frag-major over K=512
  __shared__ float  ft[16 * DMODEL];   //  8 KB, row-major
  int row0 = blockIdx.x * 16;
  for (int i = threadIdx.x; i < 16 * DMODEL; i += 256) {
    int j = i & 15, lane = (i >> 4) & 31, kbi = i >> 9;
    int m = lane & 15;
    int k = kbi * 32 + ((lane & 16) ? 8 : 0) + a_pat(j);
    xt[i] = (bf16_t)x[(row0 + m) * DMODEL + k];
  }
  __syncthreads();

  int wave = threadIdx.x >> 5, lane = threadIdx.x & 31;
  int nloc = lane & 15;
  int mo   = (lane & 16) ? 8 : 0;

  v16bf afrag[4];
#pragma unroll
  for (int kb = 0; kb < 4; ++kb) afrag[kb] = load_frag_lds(xt, kb, lane);

  // stage 1: 16x512 = 32 column tiles, each wave does 4; relu+bias, write
  // bf16 into ht at the A-fragment position for (row m, k=n).
#pragma unroll
  for (int tile = 0; tile < 4; ++tile) {
    int n0 = (wave + 8 * tile) * 16;
    v8f acc = {};
#pragma unroll
    for (int kb = 0; kb < 4; ++kb) {
      v16bf bfrag = load_frag_gl(w1, kb, wave + 8 * tile, FFDIM / 16, lane);
      acc = __builtin_amdgcn_wmma_f32_16x16x32_bf16(false, afrag[kb], false, bfrag,
                                                    (short)0, acc, false, false);
    }
    int nn   = n0 + nloc;            // K index for stage 2
    int kb2  = nn >> 5;
    int kn   = nn & 31;
    int half = (kn >> 3) & 1;
    int p    = kn - (half << 3);     // in {0..7} u {16..23}
    int jj   = (p < 8) ? p : (p - 8);
    float bias = b1[nn];
#pragma unroll
    for (int g = 0; g < 8; ++g) {
      float r = acc[g] + bias;
      int m = mo + g;
      int plane = m + (half << 4);
      ht[(((kb2 * 32) + plane) << 4) + jj] = (bf16_t)(r > 0.f ? r : 0.f);
    }
  }
  __syncthreads();

  // stage 2: 16x128, each wave one column tile, K = 512
  {
    v8f acc = {};
#pragma unroll
    for (int kb = 0; kb < 16; ++kb) {
      v16bf a2  = load_frag_lds(ht, kb, lane);
      v16bf b2f = load_frag_gl(w2, kb, wave, DMODEL / 16, lane);
      acc = __builtin_amdgcn_wmma_f32_16x16x32_bf16(false, a2, false, b2f,
                                                    (short)0, acc, false, false);
    }
    int n = wave * 16 + nloc;
    float bias = b2[n];
#pragma unroll
    for (int g = 0; g < 8; ++g)
      ft[(mo + g) * DMODEL + n] = acc[g] + bias;
  }
  __syncthreads();

  // residual + LayerNorm: 16 threads per row, each 8 columns
  int row = threadIdx.x >> 4;
  int i   = threadIdx.x & 15;
  float tv[8], s = 0.f, ss = 0.f;
#pragma unroll
  for (int c0 = 0; c0 < 8; ++c0) {
    int c = c0 * 16 + i;
    float t = x[(size_t)(row0 + row) * DMODEL + c] + ft[row * DMODEL + c];
    tv[c0] = t; s += t; ss += t * t;
  }
#pragma unroll
  for (int m = 1; m < 16; m <<= 1) { s += __shfl_xor(s, m, 32); ss += __shfl_xor(ss, m, 32); }
  float mu = s * (1.f / DMODEL);
  float var = ss * (1.f / DMODEL) - mu * mu;
  float rstd = rsqrtf(var + LN_EPS);
#pragma unroll
  for (int c0 = 0; c0 < 8; ++c0) {
    int c = c0 * 16 + i;
    x[(size_t)(row0 + row) * DMODEL + c] = (tv[c0] - mu) * rstd * g2[c] + be2[c];
  }
}

// ---------------------------------------------------------------------------
// logits = x@W_out + b_out, masked by allocation status
// ---------------------------------------------------------------------------
__global__ void out_kernel(const float* __restrict__ x, const float* __restrict__ Wout,
                           const float* __restrict__ bout, const float* __restrict__ fa,
                           const int* __restrict__ creq, float* __restrict__ out, int N) {
  int t = blockIdx.x * blockDim.x + threadIdx.x;
  if (t >= N * NFREQ) return;
  int n = t / NFREQ, f = t % NFREQ;
  float rs = 0.f;
#pragma unroll
  for (int j = 0; j < NFREQ; ++j) rs += fa[n * NFREQ + j];
  bool unalloc = rs < (float)creq[n];
  float acc = bout[f];
#pragma unroll
  for (int d = 0; d < DMODEL; ++d) acc += x[(size_t)n * DMODEL + d] * Wout[d * NFREQ + f];
  out[t] = unalloc ? acc : -__builtin_inff();
}

// ---------------------------------------------------------------------------
extern "C" void kernel_launch(void* const* d_in, const int* in_sizes, int n_in,
                              void* d_out, int out_size, void* d_ws, size_t ws_size,
                              hipStream_t stream) {
  const float* fa   = (const float*)d_in[0];
  const float* npa  = (const float*)d_in[1];
  const float* epa  = (const float*)d_in[2];
  const int*   creq = (const int*)d_in[3];
  const int*   ei   = (const int*)d_in[4];
  const float* Win  = (const float*)d_in[6];
  const float* bin  = (const float*)d_in[7];
  const float* Wemb = (const float*)d_in[8];
  const float* bemb = (const float*)d_in[9];
  const float* Wq   = (const float*)d_in[10];
  const float* bq   = (const float*)d_in[11];
  const float* Wk   = (const float*)d_in[12];
  const float* bk   = (const float*)d_in[13];
  const float* Wv   = (const float*)d_in[14];
  const float* bv   = (const float*)d_in[15];
  const float* We   = (const float*)d_in[16];
  const float* Wsk  = (const float*)d_in[17];
  const float* bsk  = (const float*)d_in[18];
  const float* W1   = (const float*)d_in[19];
  const float* b1   = (const float*)d_in[20];
  const float* W2   = (const float*)d_in[21];
  const float* b2   = (const float*)d_in[22];
  const float* g1   = (const float*)d_in[23];
  const float* be1  = (const float*)d_in[24];
  const float* g2   = (const float*)d_in[25];
  const float* be2  = (const float*)d_in[26];
  const float* Wout = (const float*)d_in[27];
  const float* bout = (const float*)d_in[28];

  const int N = in_sizes[3];
  const int E = in_sizes[4] / 2;

  // carve workspace
  char* cur = (char*)d_ws;
  auto carve = [&](size_t bytes) {
    void* p = (void*)cur;
    cur += (bytes + 255) & ~(size_t)255;
    return p;
  };
  float*    x    = (float*)carve((size_t)N * DMODEL * 4);
  float*    inp  = (float*)carve((size_t)N * DMODEL * 4);
  float*    qb   = (float*)carve((size_t)N * DMODEL * 4);
  float*    kb_  = (float*)carve((size_t)N * DMODEL * 4);
  float*    vb   = (float*)carve((size_t)N * DMODEL * 4);
  float*    msg  = (float*)carve((size_t)N * DMODEL * 4);
  float*    alp  = (float*)carve((size_t)E * NHEAD * 4);
  unsigned* mb   = (unsigned*)carve((size_t)N * NHEAD * 4);
  float*    sb   = (float*)carve((size_t)N * NHEAD * 4);
  bf16_t*   wqb  = (bf16_t*)carve((size_t)DMODEL * DMODEL * 2);
  bf16_t*   wkb  = (bf16_t*)carve((size_t)DMODEL * DMODEL * 2);
  bf16_t*   wvb  = (bf16_t*)carve((size_t)DMODEL * DMODEL * 2);
  bf16_t*   wsb  = (bf16_t*)carve((size_t)DMODEL * DMODEL * 2);
  bf16_t*   w1b  = (bf16_t*)carve((size_t)DMODEL * FFDIM * 2);
  bf16_t*   w2b  = (bf16_t*)carve((size_t)FFDIM * DMODEL * 2);
  (void)ws_size;

  const int ntiles = N / 16;            // N = 50000 -> 3125 full tiles
  const int TPB = 256;
  auto cdiv = [](int a, int b) { return (a + b - 1) / b; };

  proj_in_kernel<<<cdiv(N * DMODEL, TPB), TPB, 0, stream>>>(fa, npa, Win, bin, Wemb, bemb,
                                                            inp, x, N);

  for (int l = 0; l < NLAYER; ++l) {
    // x = x + inp
    add_kernel<<<cdiv(N * DMODEL, TPB), TPB, 0, stream>>>(x, inp, N * DMODEL);

    // per-layer weight packing: f32 row-major -> bf16 fragment-major
    const int DD = DMODEL * DMODEL, DF = DMODEL * FFDIM;
    pack_B_kernel<<<cdiv(DD, TPB), TPB, 0, stream>>>(Wq  + (size_t)l * DD, wqb, DMODEL, DMODEL);
    pack_B_kernel<<<cdiv(DD, TPB), TPB, 0, stream>>>(Wk  + (size_t)l * DD, wkb, DMODEL, DMODEL);
    pack_B_kernel<<<cdiv(DD, TPB), TPB, 0, stream>>>(Wv  + (size_t)l * DD, wvb, DMODEL, DMODEL);
    pack_B_kernel<<<cdiv(DD, TPB), TPB, 0, stream>>>(Wsk + (size_t)l * DD, wsb, DMODEL, DMODEL);
    pack_B_kernel<<<cdiv(DF, TPB), TPB, 0, stream>>>(W1  + (size_t)l * DF, w1b, DMODEL, FFDIM);
    pack_B_kernel<<<cdiv(DF, TPB), TPB, 0, stream>>>(W2  + (size_t)l * DF, w2b, FFDIM, DMODEL);

    attn_init_kernel<<<cdiv(N * NHEAD, TPB), TPB, 0, stream>>>(mb, sb, N * NHEAD);

    // q/k/v + skip (skip seeds msg buffer)
    qkvs_wmma_kernel<<<ntiles, TPB, 0, stream>>>(
        x, wqb, wkb, wvb, wsb,
        bq + (size_t)l * DMODEL, bk + (size_t)l * DMODEL,
        bv + (size_t)l * DMODEL, bsk + (size_t)l * DMODEL,
        qb, kb_, vb, msg);

    const float* Wel = We + (size_t)l * 8 * DMODEL;
    attn_alpha_kernel<<<cdiv(E * NHEAD, TPB), TPB, 0, stream>>>(qb, kb_, epa, Wel, ei, alp, mb, E);
    attn_exp_kernel<<<cdiv(E * NHEAD, TPB), TPB, 0, stream>>>(alp, mb, sb, ei, E);
    attn_msg_kernel<<<cdiv(E * NHEAD, TPB), TPB, 0, stream>>>(alp, sb, vb, epa, Wel, ei, msg, E);

    postattn_ln_kernel<<<cdiv(N, 8), TPB, 0, stream>>>(x, msg,
        g1 + (size_t)l * DMODEL, be1 + (size_t)l * DMODEL, N);

    ffn_wmma_kernel<<<ntiles, TPB, 0, stream>>>(x, w1b, b1 + (size_t)l * FFDIM,
        w2b, b2 + (size_t)l * DMODEL, g2 + (size_t)l * DMODEL, be2 + (size_t)l * DMODEL);
  }

  out_kernel<<<cdiv(N * NFREQ, TPB), TPB, 0, stream>>>(x, Wout, bout, fa, creq,
                                                       (float*)d_out, N);
}
